// TimeAwareTransformer_2826088481575
// MI455X (gfx1250) — matrix operand order repaired and checked by reading, not compiled, not run on hardware
//
#include <hip/hip_runtime.h>
#include <hip/hip_bf16.h>
#include <math.h>

// ---------------------------------------------------------------------------
// Problem constants (match reference)
// ---------------------------------------------------------------------------
#define BB 4
#define LL 128
#define DD 256
#define PP 1024
#define TU 86400
#define TB 3600
#define NEGV (-4294967295.0f)   // -2^32 + 1

typedef __attribute__((ext_vector_type(16))) __bf16 v16bf;
typedef __attribute__((ext_vector_type(8)))  float  v8f;

// Python floored division / modulo (faithful to jnp semantics)
__device__ __forceinline__ int pydiv(int a, int b) {
    int q = a / b;
    if ((a % b != 0) && ((a ^ b) < 0)) --q;
    return q;
}
__device__ __forceinline__ int pymod(int a, int b) {
    int r = a % b;
    if (r != 0 && ((r ^ b) < 0)) r += b;
    return r;
}

// ---------------------------------------------------------------------------
// Fragment loaders (f32 memory -> bf16 fragments, RNE via v_cvt_pk_bf16_f32)
//
// B fragment (32x16, K x N), this lane's column: K {p+0..7} and K {p+16..23}.
// ---------------------------------------------------------------------------
__device__ __forceinline__ v16bf load_fragB(const float* __restrict__ p) {
    const float4* p4 = (const float4*)p;
    float4 x0 = p4[0], x1 = p4[1], x2 = p4[4], x3 = p4[5];
    v16bf r;
    r[0]  = (__bf16)x0.x; r[1]  = (__bf16)x0.y; r[2]  = (__bf16)x0.z; r[3]  = (__bf16)x0.w;
    r[4]  = (__bf16)x1.x; r[5]  = (__bf16)x1.y; r[6]  = (__bf16)x1.z; r[7]  = (__bf16)x1.w;
    r[8]  = (__bf16)x2.x; r[9]  = (__bf16)x2.y; r[10] = (__bf16)x2.z; r[11] = (__bf16)x2.w;
    r[12] = (__bf16)x3.x; r[13] = (__bf16)x3.y; r[14] = (__bf16)x3.z; r[15] = (__bf16)x3.w;
    return r;
}

// A fragment (16x32, M x K): this lane needs K {p+0..7} and K {p+16..23}.
__device__ __forceinline__ v16bf load_fragA(const float* __restrict__ p) {
    const float4* p4 = (const float4*)p;
    float4 x0 = p4[0], x1 = p4[1];     // p[0..7]
    float4 x2 = p4[4], x3 = p4[5];     // p[16..23]
    v16bf r;
    r[0]  = (__bf16)x0.x; r[1]  = (__bf16)x0.y; r[2]  = (__bf16)x0.z; r[3]  = (__bf16)x0.w;
    r[4]  = (__bf16)x1.x; r[5]  = (__bf16)x1.y; r[6]  = (__bf16)x1.z; r[7]  = (__bf16)x1.w;
    r[8]  = (__bf16)x2.x; r[9]  = (__bf16)x2.y; r[10] = (__bf16)x2.z; r[11] = (__bf16)x2.w;
    r[12] = (__bf16)x3.x; r[13] = (__bf16)x3.y; r[14] = (__bf16)x3.z; r[15] = (__bf16)x3.w;
    return r;
}

// ---------------------------------------------------------------------------
// bf16-WMMA GEMM:  C[M,N] = act(A[M,KK] @ W[N,KK]^T + bias[N])
// KK compile-time (always 256 here) -> fully unrolled K loop.
// Block = 128 threads = 4 wave32's; each wave owns a 16(M) x 64(N) strip:
// one A fragment reused across 4 WMMAs per K-step (4 accumulators).
// Grid: (M/16, N/256).
//
// Fragment layouts per CDNA5 ISA 7.12.2 (wave32):
//   A (16x32 bf16): lane<16 -> row=lane,    elems = K{kb+0..7, kb+16..23}
//                   lane>=16 -> row=lane-16, elems = K{kb+8..15, kb+24..31}
//   B (32x16 bf16): lane<16 -> col=lane,    elems = K{kb+0..7, kb+16..23}
//                   lane>=16 -> col=lane-16, elems = K{kb+8..15, kb+24..31}
//     (B[k][n] = W[n][k] -> two 8-float runs of the weight row)
//   C/D (16x16 f32): lane<16 -> N=lane, M=v ; lane>=16 -> N=lane-16, M=v+8
// ---------------------------------------------------------------------------
template<int KK>
__global__ void gemm_bf16_wmma(const float* __restrict__ A,
                               const float* __restrict__ W,
                               const float* __restrict__ bias,
                               float* __restrict__ C,
                               int M, int N, int relu)
{
    const int lane  = threadIdx.x & 31;
    const int wave  = threadIdx.x >> 5;
    const int mbase = blockIdx.x * 16;
    const int nwave = blockIdx.y * 256 + wave * 64;   // 4 x 16 columns per wave
    const int l15   = lane & 15;
    const int hi    = lane >> 4;          // 0 for lanes 0-15, 1 for 16-31
    const int row   = mbase + l15;

    const float* __restrict__ arow = A + (size_t)row * KK;
    const float* __restrict__ w0   = W + (size_t)(nwave +  0 + l15) * KK;
    const float* __restrict__ w1   = W + (size_t)(nwave + 16 + l15) * KK;
    const float* __restrict__ w2   = W + (size_t)(nwave + 32 + l15) * KK;
    const float* __restrict__ w3   = W + (size_t)(nwave + 48 + l15) * KK;

    __builtin_prefetch(arow, 0, 1);
    __builtin_prefetch(w0, 0, 1);
    __builtin_prefetch(w2, 0, 1);

    v8f acc0 = {}, acc1 = {}, acc2 = {}, acc3 = {};

#pragma unroll
    for (int kb = 0; kb < KK; kb += 32) {
        const int ka = kb + hi * 8;       // A-fragment K base for this lane
        const int kc = kb + hi * 8;       // B-fragment K base for this lane
        const v16bf af = load_fragA(arow + ka);
        const v16bf b0 = load_fragB(w0 + kc);
        const v16bf b1 = load_fragB(w1 + kc);
        const v16bf b2 = load_fragB(w2 + kc);
        const v16bf b3 = load_fragB(w3 + kc);
        acc0 = __builtin_amdgcn_wmma_f32_16x16x32_bf16(false, af, false, b0,
                                                       (short)0, acc0, false, false);
        acc1 = __builtin_amdgcn_wmma_f32_16x16x32_bf16(false, af, false, b1,
                                                       (short)0, acc1, false, false);
        acc2 = __builtin_amdgcn_wmma_f32_16x16x32_bf16(false, af, false, b2,
                                                       (short)0, acc2, false, false);
        acc3 = __builtin_amdgcn_wmma_f32_16x16x32_bf16(false, af, false, b3,
                                                       (short)0, acc3, false, false);
    }

    // Epilogue: bias (+ optional ReLU), scatter per C/D fragment layout.
#pragma unroll
    for (int sub = 0; sub < 4; ++sub) {
        const v8f acc = (sub == 0) ? acc0 : (sub == 1) ? acc1 : (sub == 2) ? acc2 : acc3;
        const int col = nwave + sub * 16 + l15;
        const float bvv = bias ? bias[col] : 0.0f;
#pragma unroll
        for (int v = 0; v < 8; ++v) {
            const int m = mbase + v + hi * 8;
            float val = acc[v] + bvv;
            if (relu) val = fmaxf(val, 0.0f);
            C[(size_t)m * N + col] = val;
        }
    }
}

// ---------------------------------------------------------------------------
// Fused attention per (b,j): scores (with time-interval embedding terms),
// masked softmax, weighted sum over (V + hE + dE), residual + LayerNorm.
// One block of 256 threads per (b,j). All gathers hit L1/L2-resident tables.
// ---------------------------------------------------------------------------
__global__ void attn_kernel(const float* __restrict__ Q,
                            const float* __restrict__ Km,
                            const float* __restrict__ V,
                            const float* __restrict__ src,
                            const float* __restrict__ hour_emb,
                            const float* __restrict__ day_emb,
                            const float* __restrict__ g11,
                            const float* __restrict__ b11,
                            const int* __restrict__ seq_lens,
                            const int* __restrict__ ts,
                            float* __restrict__ X)
{
    const int bj = blockIdx.x;
    const int b  = bj / LL;
    const int j  = bj % LL;
    const int t  = threadIdx.x;            // 0..255

    __shared__ float sQ[DD];
    __shared__ float s_aw[LL];
    __shared__ int   s_h[LL];
    __shared__ int   s_d[LL];
    __shared__ float red[256];

    const bool jvalid = (j < seq_lens[b]);

    sQ[t] = Q[(size_t)(b * LL + j) * DD + t];
    __syncthreads();

    // ---- phase 1: indices + scores, one k per thread (t < 128) ----
    if (t < LL) {
        const int k = t;
        const bool valid = (k <= j) && jvalid;
        const int diff = ts[b * LL + j] - ts[b * LL + k];
        // faithful quirk: batch index compared against row index j
        const int hour = valid ? ((b == j) ? 1 : (pymod(diff, TU) / TB + 2)) : 0;
        const int day  = valid ? min(pydiv(diff, TU) + 1, 7) : 0;
        s_h[k] = hour;
        s_d[k] = day;

        float sc;
        if (valid) {
            const float* kr = Km + (size_t)(b * LL + k) * DD;
            const float* he = hour_emb + (size_t)hour * DD;
            const float* de = day_emb  + (size_t)day  * DD;
            float acc = 0.0f;
            for (int d = 0; d < DD; ++d)
                acc = fmaf(sQ[d], kr[d] + he[d] + de[d], acc);
            sc = acc * 0.0625f;            // 1/sqrt(256)
        } else {
            sc = NEGV;
        }
        s_aw[k] = sc;
    }
    __syncthreads();

    // ---- phase 2: softmax over k (all-NEG rows -> uniform, as in ref) ----
    red[t] = (t < LL) ? s_aw[t] : -INFINITY;
    __syncthreads();
    for (int s = 128; s > 0; s >>= 1) {
        if (t < s) red[t] = fmaxf(red[t], red[t + s]);
        __syncthreads();
    }
    const float mx = red[0];
    __syncthreads();

    float e = 0.0f;
    if (t < LL) e = __expf(s_aw[t] - mx);
    red[t] = (t < LL) ? e : 0.0f;
    __syncthreads();
    for (int s = 128; s > 0; s >>= 1) {
        if (t < s) red[t] += red[t + s];
        __syncthreads();
    }
    const float inv_sum = 1.0f / red[0];
    __syncthreads();
    if (t < LL) s_aw[t] = e * inv_sum;
    __syncthreads();

    // ---- phase 3: x[d] = sum_k aw[k] * (V[k,d] + hE[d] + dE[d]) ----
    float acc = 0.0f;
    for (int k = 0; k < LL; ++k) {
        const float w = s_aw[k];
        acc = fmaf(w,
                   V[(size_t)(b * LL + k) * DD + t]
                   + hour_emb[(size_t)s_h[k] * DD + t]
                   + day_emb[(size_t)s_d[k] * DD + t],
                   acc);
    }
    const float y = acc + src[(size_t)(b * LL + j) * DD + t];

    // ---- LayerNorm (population variance, eps 1e-5) ----
    red[t] = y;
    __syncthreads();
    for (int s = 128; s > 0; s >>= 1) {
        if (t < s) red[t] += red[t + s];
        __syncthreads();
    }
    const float mean = red[0] * (1.0f / DD);
    __syncthreads();
    const float dev = y - mean;
    red[t] = dev * dev;
    __syncthreads();
    for (int s = 128; s > 0; s >>= 1) {
        if (t < s) red[t] += red[t + s];
        __syncthreads();
    }
    const float var = red[0] * (1.0f / DD);
    X[(size_t)(b * LL + j) * DD + t] =
        dev * rsqrtf(var + 1e-5f) * g11[t] + b11[t];
}

// ---------------------------------------------------------------------------
// LayerNorm of (X + F2) -> F, one block of 256 threads per row
// ---------------------------------------------------------------------------
__global__ void ln2_kernel(const float* __restrict__ X,
                           const float* __restrict__ F2,
                           const float* __restrict__ g12,
                           const float* __restrict__ b12,
                           float* __restrict__ F)
{
    const int r = blockIdx.x;
    const int t = threadIdx.x;
    __shared__ float red[256];

    const float y = X[(size_t)r * DD + t] + F2[(size_t)r * DD + t];
    red[t] = y;
    __syncthreads();
    for (int s = 128; s > 0; s >>= 1) {
        if (t < s) red[t] += red[t + s];
        __syncthreads();
    }
    const float mean = red[0] * (1.0f / DD);
    __syncthreads();
    const float dev = y - mean;
    red[t] = dev * dev;
    __syncthreads();
    for (int s = 128; s > 0; s >>= 1) {
        if (t < s) red[t] += red[t + s];
        __syncthreads();
    }
    const float var = red[0] * (1.0f / DD);
    F[(size_t)r * DD + t] = dev * rsqrtf(var + 1e-5f) * g12[t] + b12[t];
}

// ---------------------------------------------------------------------------
// Pool pre-pass: u[b,j] = (sum_{k<=j} f[b,k] + lhE[b,j,k] + ldE[b,j,k]) / (j+1)
// (decoder matmul + bdec are hoisted out; tril sum commutes with the GEMM)
// ---------------------------------------------------------------------------
__global__ void poolpre_kernel(const float* __restrict__ F,
                               const float* __restrict__ hour_emb,
                               const float* __restrict__ day_emb,
                               const int* __restrict__ seq_lens,
                               const int* __restrict__ ts,
                               const int* __restrict__ lts,
                               float* __restrict__ U)
{
    const int bj = blockIdx.x;
    const int b  = bj / LL;
    const int j  = bj % LL;
    const int t  = threadIdx.x;            // 0..255

    __shared__ int s_h[LL];
    __shared__ int s_d[LL];

    const bool jvalid = (j < seq_lens[b]);
    if (t < LL) {
        const int k = t;
        const bool valid = (k <= j) && jvalid;
        const int ldiff = lts[b * LL + j] - ts[b * LL + k];
        s_h[k] = valid ? (pymod(ldiff, TU) / TB + 2) : 0;
        s_d[k] = valid ? min(pydiv(ldiff, TU) + 1, 7) : 0;
    }
    __syncthreads();

    float acc = 0.0f;
    for (int k = 0; k <= j; ++k) {
        acc += F[(size_t)(b * LL + k) * DD + t]
             + hour_emb[(size_t)s_h[k] * DD + t]
             + day_emb[(size_t)s_d[k] * DD + t];
    }
    U[(size_t)(b * LL + j) * DD + t] = acc / (float)(j + 1);
}

// ---------------------------------------------------------------------------
// Host-side launch
// ---------------------------------------------------------------------------
extern "C" void kernel_launch(void* const* d_in, const int* in_sizes, int n_in,
                              void* d_out, int out_size, void* d_ws, size_t ws_size,
                              hipStream_t stream)
{
    (void)in_sizes; (void)n_in; (void)out_size; (void)ws_size;

    const float* src      = (const float*)d_in[0];
    const float* hour_emb = (const float*)d_in[1];
    const float* day_emb  = (const float*)d_in[2];
    const float* wq       = (const float*)d_in[3];
    const float* bq       = (const float*)d_in[4];
    const float* wk       = (const float*)d_in[5];
    const float* bk       = (const float*)d_in[6];
    const float* wv       = (const float*)d_in[7];
    const float* bv       = (const float*)d_in[8];
    const float* g11      = (const float*)d_in[9];
    const float* b11      = (const float*)d_in[10];
    const float* wf1      = (const float*)d_in[11];
    const float* bf1      = (const float*)d_in[12];
    const float* wf2      = (const float*)d_in[13];
    const float* bf2      = (const float*)d_in[14];
    const float* g12      = (const float*)d_in[15];
    const float* b12      = (const float*)d_in[16];
    const float* wdec     = (const float*)d_in[17];
    const float* bdec     = (const float*)d_in[18];
    const int*   slen     = (const int*)d_in[19];
    const int*   ts       = (const int*)d_in[20];
    const int*   lts      = (const int*)d_in[21];
    float*       out      = (float*)d_out;

    const int M = BB * LL;                // 512 rows
    const size_t RS = (size_t)M * DD;     // elements per [512,256] buffer

    float* ws = (float*)d_ws;
    float* Q  = ws + 0 * RS;
    float* K  = ws + 1 * RS;
    float* V  = ws + 2 * RS;
    float* X  = ws + 3 * RS;
    float* F1 = ws + 4 * RS;
    float* F2 = ws + 5 * RS;
    float* F  = ws + 6 * RS;
    float* U  = ws + 7 * RS;

    const dim3 gD(M / 16, DD / 256);      // GEMMs with N = 256 -> grid.y = 1
    const dim3 gP(M / 16, PP / 256);      // decoder GEMM, N = 1024 -> grid.y = 4

    // QKV projections (bf16 WMMA, f32 accumulate)
    gemm_bf16_wmma<DD><<<gD, 128, 0, stream>>>(src, wq, bq, Q, M, DD, 0);
    gemm_bf16_wmma<DD><<<gD, 128, 0, stream>>>(src, wk, bk, K, M, DD, 0);
    gemm_bf16_wmma<DD><<<gD, 128, 0, stream>>>(src, wv, bv, V, M, DD, 0);

    // Fused attention + residual + LN1
    attn_kernel<<<M, 256, 0, stream>>>(Q, K, V, src, hour_emb, day_emb,
                                       g11, b11, slen, ts, X);

    // FFN (ReLU on first GEMM) + residual LN2
    gemm_bf16_wmma<DD><<<gD, 128, 0, stream>>>(X,  wf1, bf1, F1, M, DD, 1);
    gemm_bf16_wmma<DD><<<gD, 128, 0, stream>>>(F1, wf2, bf2, F2, M, DD, 0);
    ln2_kernel<<<M, 256, 0, stream>>>(X, F2, g12, b12, F);

    // Tril-pooled pre-sum, then decoder GEMM directly into d_out
    poolpre_kernel<<<M, 256, 0, stream>>>(F, hour_emb, day_emb, slen, ts, lts, U);
    gemm_bf16_wmma<DD><<<gP, 128, 0, stream>>>(U, wdec, bdec, out, M, PP, 0);
}
